// Encoder_39865886442296
// MI455X (gfx1250) — matrix operand order repaired and checked by reading, not compiled, not run on hardware
//
#include <hip/hip_runtime.h>

// ---------------------------------------------------------------------------
// 2-layer GCN for MI455X (gfx1250, wave32).
// GEMMs: v_wmma_f32_16x16x32_bf16 (fp32 accumulate), with weights pre-packed
// into WMMA-fragment-major layout so each lane's B fragment is one 32-byte
// contiguous vector load. Edge aggregation: wave-per-edge gather +
// global_atomic_add_f32 scatter (L2-resident: h, edge_index << 192MB L2).
// ---------------------------------------------------------------------------

typedef __attribute__((ext_vector_type(16))) __bf16 v16bf;
typedef __attribute__((ext_vector_type(8)))  float  v8f;

#define NNODES   100000
#define IN_DIM   256
#define H1_DIM   128
#define H2_DIM   64

// --- WMMA A-fragment K index (16-bit A 16x32, ISA 7.12.2) -------------------
// VGPR v<4: K = 8*half + 2v (+lo);  v>=4: K = 16 + 8*half + 2(v-4) (+lo)
__device__ __forceinline__ int kmap_a(int j, int half) {
    int v  = j >> 1;
    int lo = j & 1;
    int k  = (v < 4) ? (half * 8 + (v << 1))
                     : (16 + half * 8 + ((v - 4) << 1));
    return k + lo;
}

// --- pack W[K,NOUT] f32 -> WMMA B-fragment-major bf16 -----------------------
// Fragment f = (nTile*(K/32) + kTile)*32 + lane holds the 16 bf16 values
// B[kTile*32 + half*16 + j, nTile*16 + (lane&15)], j = 0..15, contiguously.
template <int K, int NOUT>
__global__ void k_pack_b(const float* __restrict__ W, __bf16* __restrict__ Bpk) {
    const int total = (NOUT / 16) * (K / 32) * 32 * 16;
    int idx = blockIdx.x * blockDim.x + threadIdx.x;
    if (idx >= total) return;
    const int j    = idx & 15;
    const int lane = (idx >> 4) & 31;
    const int rest = idx >> 9;
    const int t    = rest % (K / 32);      // k-tile (compile-time pow2 divisor)
    const int nt   = rest / (K / 32);      // n-tile
    const int half = lane >> 4;
    const int n    = nt * 16 + (lane & 15);
    const int k    = t * 32 + half * 16 + j;
    Bpk[idx] = (__bf16)W[k * NOUT + n];
}

// --- GEMM: C[M,NOUT] = A[M,K] x Bpk, one wave per 16x16 C tile --------------
template <int K, int NOUT, typename AT>
__global__ void k_gemm_wmma(const AT* __restrict__ A,
                            const __bf16* __restrict__ Bpk,
                            float* __restrict__ C) {
    const int lane  = threadIdx.x;      // 0..31, no divergence (EXEC all ones)
    const int half  = lane >> 4;
    const int mn    = lane & 15;
    const int mBase = blockIdx.x << 4;
    const int nBase = blockIdx.y << 4;

    v8f acc = {};
    const AT* __restrict__ arow = A + (size_t)(mBase + mn) * K;
    // packed B fragments for this n-tile, this lane
    const v16bf* __restrict__ bp =
        (const v16bf*)Bpk + (size_t)blockIdx.y * (K / 32) * 32 + lane;

#pragma unroll
    for (int t = 0; t < K / 32; ++t) {
        v16bf af;
#pragma unroll
        for (int j = 0; j < 16; ++j)    // one contiguous 16-elem run per lane
            af[j] = (__bf16)(float)arow[t * 32 + half * 16 + j];
        v16bf bfrag = bp[t * 32];       // 32B contiguous -> global_load_b128 x2
        acc = __builtin_amdgcn_wmma_f32_16x16x32_bf16(
            /*neg_a=*/false, af, /*neg_b=*/false, bfrag,
            /*c_mod=*/(short)0, acc, /*reuse_a=*/false, /*reuse_b=*/false);
    }

#pragma unroll
    for (int r = 0; r < 8; ++r)   // C/D: VGPR r -> row r + 8*half, col = mn
        C[(size_t)(mBase + r + half * 8) * NOUT + nBase + mn] = acc[r];
}

// --- small helpers ----------------------------------------------------------
__global__ void k_deg_init(float* __restrict__ deg, int n) {
    int i = blockIdx.x * blockDim.x + threadIdx.x;
    if (i < n) deg[i] = 1.0f;                       // self-loop contribution
}

__global__ void k_deg_edges(const long long* __restrict__ row,
                            float* __restrict__ deg, long long e) {
    long long i = (long long)blockIdx.x * blockDim.x + threadIdx.x;
    if (i < e) atomicAdd(&deg[(int)row[i]], 1.0f);
}

__global__ void k_dinv(const float* __restrict__ deg,
                       float* __restrict__ dinv, int n) {
    int i = blockIdx.x * blockDim.x + threadIdx.x;
    if (i < n) dinv[i] = __frsqrt_rn(deg[i]);       // deg >= 1 always
}

// agg[i,:] = h[i,:] * dinv[i]^2   (self-loop term; also initializes agg)
template <int H, int SH>
__global__ void k_selfloop_init(const float* __restrict__ h,
                                const float* __restrict__ dinv,
                                float* __restrict__ agg, long long total) {
    long long i = (long long)blockIdx.x * blockDim.x + threadIdx.x;
    if (i < total) {
        float di = dinv[(int)(i >> SH)];
        agg[i] = h[i] * di * di;
    }
}

// one wave per edge: agg[row,:] += h[col,:] * (dinv[row]*dinv[col])
template <int H>
__global__ void k_edge_scatter(const long long* __restrict__ row,
                               const long long* __restrict__ col,
                               const float* __restrict__ dinv,
                               const float* __restrict__ h,
                               float* __restrict__ agg, long long e) {
    const int lane = threadIdx.x & 31;
    long long w = (long long)blockIdx.x * (blockDim.x >> 5) + (threadIdx.x >> 5);
    if (w >= e) return;
    const int r = (int)row[w];
    const int c = (int)col[w];
    const float nrm = dinv[r] * dinv[c];
    const float* __restrict__ hs = h + (size_t)c * H;
    float* __restrict__ ad = agg + (size_t)r * H;
#pragma unroll
    for (int ch = lane; ch < H; ch += 32)
        atomicAdd(&ad[ch], hs[ch] * nrm);           // global_atomic_add_f32
}

// out_bf16 = relu(agg + bias)   (layer-1 activation, feeds bf16 GEMM2)
template <int H>
__global__ void k_relu_bias_bf16(const float* __restrict__ agg,
                                 const float* __restrict__ bias,
                                 __bf16* __restrict__ out, long long total) {
    long long i = (long long)blockIdx.x * blockDim.x + threadIdx.x;
    if (i < total) {
        float v = agg[i] + bias[(int)(i & (H - 1))];
        out[i] = (__bf16)fmaxf(v, 0.0f);
    }
}

// y = relu(y + bias) in place   (final output)
template <int H>
__global__ void k_relu_bias_f32(float* __restrict__ y,
                                const float* __restrict__ bias,
                                long long total) {
    long long i = (long long)blockIdx.x * blockDim.x + threadIdx.x;
    if (i < total) {
        float v = y[i] + bias[(int)(i & (H - 1))];
        y[i] = fmaxf(v, 0.0f);
    }
}

// ---------------------------------------------------------------------------
extern "C" void kernel_launch(void* const* d_in, const int* in_sizes, int n_in,
                              void* d_out, int out_size, void* d_ws, size_t ws_size,
                              hipStream_t stream) {
    const float*     x  = (const float*)d_in[0];
    const long long* ei = (const long long*)d_in[1];   // [2, E] int64
    const float*     W1 = (const float*)d_in[2];
    const float*     b1 = (const float*)d_in[3];
    const float*     W2 = (const float*)d_in[4];
    const float*     b2 = (const float*)d_in[5];
    float*           out = (float*)d_out;

    const int       N = in_sizes[0] / IN_DIM;          // 100000
    const long long E = (long long)in_sizes[1] / 2;    // 3200000
    const long long* row = ei;
    const long long* col = ei + E;

    // workspace layout (~103.3 MB, all offsets 32B-aligned)
    char* ws = (char*)d_ws;
    float*  agg1  = (float*)ws;                                    // N*H1*4
    char*   bufB  = ws + (size_t)N * H1_DIM * 4;                   // N*H1*4
    float*  h1    = (float*)bufB;                                  //  fp32 h1
    __bf16* h1bf  = (__bf16*)bufB;                                 //  later: bf16 relu(agg1)
    float*  h2    = (float*)(bufB + (size_t)N * H1_DIM * 2);       //  later: fp32 h2
    char*   tail  = bufB + (size_t)N * H1_DIM * 4;
    float*  deg   = (float*)tail;
    float*  dinv  = deg + N;
    __bf16* W1pk  = (__bf16*)(dinv + N);                           // packed fragments
    __bf16* W2pk  = W1pk + IN_DIM * H1_DIM;

    const int TPB = 256;
    const int warpsPerBlk = TPB / 32;

    // 0) weights -> bf16 WMMA-fragment-major packing
    k_pack_b<IN_DIM, H1_DIM><<<(IN_DIM * H1_DIM + TPB - 1) / TPB, TPB, 0, stream>>>(W1, W1pk);
    k_pack_b<H1_DIM, H2_DIM><<<(H1_DIM * H2_DIM + TPB - 1) / TPB, TPB, 0, stream>>>(W2, W2pk);

    // 1) degree + dinv (shared by both layers)
    k_deg_init<<<(N + TPB - 1) / TPB, TPB, 0, stream>>>(deg, N);
    k_deg_edges<<<(unsigned)((E + TPB - 1) / TPB), TPB, 0, stream>>>(row, deg, E);
    k_dinv<<<(N + TPB - 1) / TPB, TPB, 0, stream>>>(deg, dinv, N);

    // 2) layer 1: h1 = x @ W1  (WMMA bf16, 6250 x 8 tiles)
    k_gemm_wmma<IN_DIM, H1_DIM, float>
        <<<dim3(N / 16, H1_DIM / 16), 32, 0, stream>>>(x, W1pk, h1);

    // 3) aggregate: agg1 = D^-1/2 (A+I) D^-1/2 h1
    const long long tot1 = (long long)N * H1_DIM;
    k_selfloop_init<H1_DIM, 7><<<(unsigned)((tot1 + TPB - 1) / TPB), TPB, 0, stream>>>(h1, dinv, agg1, tot1);
    k_edge_scatter<H1_DIM><<<(unsigned)((E + warpsPerBlk - 1) / warpsPerBlk), TPB, 0, stream>>>(
        row, col, dinv, h1, agg1, E);

    // 4) relu(+b1) -> bf16 (h1 fp32 is dead; reuse its buffer)
    k_relu_bias_bf16<H1_DIM><<<(unsigned)((tot1 + TPB - 1) / TPB), TPB, 0, stream>>>(agg1, b1, h1bf, tot1);

    // 5) layer 2: h2 = h1bf @ W2  (WMMA bf16, 6250 x 4 tiles)
    k_gemm_wmma<H1_DIM, H2_DIM, __bf16>
        <<<dim3(N / 16, H2_DIM / 16), 32, 0, stream>>>(h1bf, W2pk, h2);

    // 6) aggregate straight into d_out, then relu(+b2) in place
    const long long tot2 = (long long)N * H2_DIM;
    k_selfloop_init<H2_DIM, 6><<<(unsigned)((tot2 + TPB - 1) / TPB), TPB, 0, stream>>>(h2, dinv, out, tot2);
    k_edge_scatter<H2_DIM><<<(unsigned)((E + warpsPerBlk - 1) / warpsPerBlk), TPB, 0, stream>>>(
        row, col, dinv, h2, out, E);
    k_relu_bias_f32<H2_DIM><<<(unsigned)((tot2 + TPB - 1) / TPB), TPB, 0, stream>>>(out, b2, tot2);
}
